// DenseKVMemory_87943750353224
// MI455X (gfx1250) — matrix operand order repaired and checked by reading, not compiled, not run on hardware
//
#include <hip/hip_runtime.h>

// DenseKVMemory for MI455X (gfx1250, wave32).
//  Pass 0: ring-buffer memory update (pure bandwidth, float4).
//  Pass 1: K/V -> bf16 WMMA-fragment pre-swizzle into d_ws (keep factor baked in).
//  Pass 2: key-split flash attention, v_wmma_f32_16x16x32_bf16, transposed logits,
//          double-buffered fragment loads, v_perm_b32 packing; partials to d_ws.
//  Pass 3: log-sum-exp reduction of 16 partials -> y.
// Fallback (small ws): self-contained single-pass flash attention.

#define BM 32768   // memory slots
#define SS 512     // segment length
#define NH 8       // heads
#define HD 64      // head dim

#define NTILE  ((BM + SS) / 16)   // 2080 16-key tiles
#define NCHUNK ((BM + SS) / 32)   // 1040 32-key chunks
#define NSPLIT 16
#define CPS    (NCHUNK / NSPLIT)  // 65 chunks per split
#define SEGC   (BM / 32)          // first chunk containing segment keys

typedef __attribute__((ext_vector_type(16))) __bf16          v16bf;
typedef __attribute__((ext_vector_type(16))) unsigned short  v16us;
typedef __attribute__((ext_vector_type(8)))  unsigned int    v8u;
typedef __attribute__((ext_vector_type(8)))  float           v8f;

__device__ __forceinline__ unsigned short f2bf(float f) {
  unsigned int u = __float_as_uint(f);
  u += 0x7FFFu + ((u >> 16) & 1u);   // round-to-nearest-even
  return (unsigned short)(u >> 16);
}
// f32 bits with bf16 RNE rounding applied (result lives in bits [31:16]).
__device__ __forceinline__ unsigned int rne_bits(float f) {
  unsigned int u = __float_as_uint(f);
  return u + 0x7FFFu + ((u >> 16) & 1u);
}

// ---------------------------------------------------------------------------
// Pass 0: new_mem = keep*mem with segment KV written at clamped write_index.
// ---------------------------------------------------------------------------
__global__ __launch_bounds__(256)
void memupd_kernel(const float* __restrict__ memK, const float* __restrict__ memV,
                   const float* __restrict__ segK, const float* __restrict__ segV,
                   const unsigned char* __restrict__ sos, const int* __restrict__ wip,
                   float* __restrict__ outK, float* __restrict__ outV,
                   float* __restrict__ outWI)
{
  const long long idx   = (long long)blockIdx.x * blockDim.x + threadIdx.x;
  const long long total = (long long)BM * NH * HD / 4;
  const int wi = wip[0];
  if (idx == 0) outWI[0] = (float)((wi + SS) % BM);
  if (idx >= total) return;

  const float keep = sos[0] ? 0.0f : 1.0f;
  int wic = wi; if (wic < 0) wic = 0; if (wic > BM - SS) wic = BM - SS;
  const int m   = (int)(idx >> 7);
  const int off = ((int)idx & 127) * 4;

  float4 k4, v4;
  if (m >= wic && m < wic + SS) {
    const size_t s = (size_t)(m - wic) * NH * HD + off;
    k4 = *(const float4*)(segK + s);
    v4 = *(const float4*)(segV + s);
  } else {
    const size_t s = (size_t)m * NH * HD + off;
    k4 = *(const float4*)(memK + s);
    v4 = *(const float4*)(memV + s);
    k4.x *= keep; k4.y *= keep; k4.z *= keep; k4.w *= keep;
    v4.x *= keep; v4.y *= keep; v4.z *= keep; v4.w *= keep;
  }
  const size_t d = (size_t)m * NH * HD + off;
  *(float4*)(outK + d) = k4;
  *(float4*)(outV + d) = v4;
}

// ---------------------------------------------------------------------------
// Pass 1a: K -> bf16 A-fragment stream. One wave per (h, tile, cc).
// ---------------------------------------------------------------------------
__global__ __launch_bounds__(256)
void prepk_kernel(const float* __restrict__ memK, const float* __restrict__ segK,
                  const unsigned char* __restrict__ sos,
                  unsigned short* __restrict__ wsK)
{
  const int w = blockIdx.x * 8 + (threadIdx.x >> 5);
  if (w >= NH * NTILE * 2) return;
  const int lane = threadIdx.x & 31;
  const int n    = lane & 15;
  const int half = lane >> 4;
  const int cc   = w & 1;
  const int t    = (w >> 1) % NTILE;
  const int h    = (w >> 1) / NTILE;
  const float keep = sos[0] ? 0.0f : 1.0f;

  const int key = t * 16 + n;
  const bool seg = key >= BM;
  const float ksc = seg ? 1.0f : keep;
  const float* kr = seg ? (segK + ((size_t)(key - BM) * NH + h) * HD)
                        : (memK + ((size_t)key       * NH + h) * HD);
  const int d0 = cc * 32 + half * 8;
  float4 a0 = *(const float4*)(kr + d0);
  float4 a1 = *(const float4*)(kr + d0 + 4);
  float4 a2 = *(const float4*)(kr + d0 + 16);
  float4 a3 = *(const float4*)(kr + d0 + 20);
  v16us u;
  u[0]=f2bf(a0.x*ksc); u[1]=f2bf(a0.y*ksc); u[2]=f2bf(a0.z*ksc); u[3]=f2bf(a0.w*ksc);
  u[4]=f2bf(a1.x*ksc); u[5]=f2bf(a1.y*ksc); u[6]=f2bf(a1.z*ksc); u[7]=f2bf(a1.w*ksc);
  u[8]=f2bf(a2.x*ksc); u[9]=f2bf(a2.y*ksc); u[10]=f2bf(a2.z*ksc); u[11]=f2bf(a2.w*ksc);
  u[12]=f2bf(a3.x*ksc); u[13]=f2bf(a3.y*ksc); u[14]=f2bf(a3.z*ksc); u[15]=f2bf(a3.w*ksc);
  *(v16us*)(wsK + (size_t)w * 512 + lane * 16) = u;
}

// ---------------------------------------------------------------------------
// Pass 1b: V -> bf16 V^T A-fragment stream. One wave per (h, chunk); g-loop.
// ---------------------------------------------------------------------------
__global__ __launch_bounds__(256)
void prepv_kernel(const float* __restrict__ memV, const float* __restrict__ segV,
                  const unsigned char* __restrict__ sos,
                  unsigned short* __restrict__ wsV)
{
  const int w = blockIdx.x * 8 + (threadIdx.x >> 5);
  if (w >= NH * NCHUNK) return;
  const int lane = threadIdx.x & 31;
  const int n    = lane & 15;
  const int half = lane >> 4;
  const int c    = w % NCHUNK;
  const int h    = w / NCHUNK;
  const float keep = sos[0] ? 0.0f : 1.0f;

  const float* rowp[16];
  float rs[16];
  #pragma unroll
  for (int j = 0; j < 16; ++j) {
    const int kmap = half * 16 + j;
    const int key  = c * 32 + kmap;
    const bool seg = key >= BM;
    rs[j]   = seg ? 1.0f : keep;
    rowp[j] = seg ? (segV + ((size_t)(key - BM) * NH + h) * HD)
                  : (memV + ((size_t)key       * NH + h) * HD);
  }
  #pragma unroll
  for (int g = 0; g < 4; ++g) {
    const int d = g * 16 + n;
    v16us u;
    #pragma unroll
    for (int j = 0; j < 16; ++j) u[j] = f2bf(rowp[j][d] * rs[j]);
    *(v16us*)(wsV + ((size_t)w * 4 + g) * 512 + lane * 16) = u;
  }
}

// ---------------------------------------------------------------------------
// Pass 2: key-split flash attention, double-buffered fragment loads.
// ---------------------------------------------------------------------------
__global__ __launch_bounds__(32)
void attn_split_kernel(const unsigned short* __restrict__ wsK,
                       const unsigned short* __restrict__ wsV,
                       const float* __restrict__ Q,
                       float* __restrict__ wsO, float* __restrict__ wsM,
                       float* __restrict__ wsL)
{
  const int lane  = threadIdx.x;
  const int n     = lane & 15;
  const int half  = lane >> 4;
  const int split = blockIdx.x & (NSPLIT - 1);
  const int qt    = (blockIdx.x >> 4) & 31;
  const int h     = blockIdx.x >> 9;
  const int q0    = qt * 16;
  const float scale = 0.125f;                       // 1/sqrt(64)
  const unsigned int selw = half ? 0x07060302u : 0x05040100u; // v_perm selector

  // Q^T B-fragments: b[j] = Q[q0+n][c*32 + 8*half + (j<8 ? j : 8+j)] * scale
  v16bf qb[2];
  {
    const float* qr = Q + ((size_t)(q0 + n) * NH + h) * HD;
    #pragma unroll
    for (int cfr = 0; cfr < 2; ++cfr) {
      const int d0 = cfr * 32 + half * 8;
      v16us u;
      #pragma unroll
      for (int i = 0; i < 4; ++i) {
        float4 v = *(const float4*)(qr + d0 + (i & 1) * 4 + (i >> 1) * 16);
        u[i*4+0] = f2bf(v.x * scale); u[i*4+1] = f2bf(v.y * scale);
        u[i*4+2] = f2bf(v.z * scale); u[i*4+3] = f2bf(v.w * scale);
      }
      qb[cfr] = __builtin_bit_cast(v16bf, u);
    }
  }

  float m_run = -3.0e38f, l_run = 0.0f;
  v8f acc[4] = { v8f{}, v8f{}, v8f{}, v8f{} };

  const size_t laneOff = (size_t)lane * 16;
  const int c_lo = split * CPS;

  // Fragment loads for one chunk: 4 K-frags (t2*2+cc) + 4 V-frags (g), 32B each.
  auto loadF = [&](int c, v16us* kf, v16us* vf) {
    const unsigned short* kB = wsK + ((size_t)(h * NTILE + c * 2) * 2) * 512;
    const unsigned short* vB = wsV + ((size_t)(h * NCHUNK + c) * 4) * 512;
    #pragma unroll
    for (int i = 0; i < 4; ++i) kf[i] = *(const v16us*)(kB + (size_t)i * 512 + laneOff);
    #pragma unroll
    for (int g = 0; g < 4; ++g) vf[g] = *(const v16us*)(vB + (size_t)g * 512 + laneOff);
  };

  auto computeC = [&](int c, const v16us* kf, const v16us* vf) {
    // ---- ST[key][query] = Ktile x Q^T (two 16-key tiles) ----
    float st[2][8];
    #pragma unroll
    for (int t2 = 0; t2 < 2; ++t2) {
      v8f cacc = v8f{};
      #pragma unroll
      for (int cc = 0; cc < 2; ++cc)
        cacc = __builtin_amdgcn_wmma_f32_16x16x32_bf16(
                   false, __builtin_bit_cast(v16bf, kf[t2 * 2 + cc]), false, qb[cc],
                   (short)0, cacc, false, false);
      #pragma unroll
      for (int r = 0; r < 8; ++r) st[t2][r] = cacc[r];
    }

    // ---- causal mask (segment keys only) ----
    if (c >= SEGC) {
      #pragma unroll
      for (int t2 = 0; t2 < 2; ++t2)
        #pragma unroll
        for (int r = 0; r < 8; ++r) {
          const int key = c * 32 + t2 * 16 + r + 8 * half;
          if (key >= BM && (key - BM) >= q0 + n) st[t2][r] = -3.0e38f;
        }
    }

    // ---- online softmax (per query column; lanes l and l+16 agree) ----
    float tmax = st[0][0];
    #pragma unroll
    for (int r = 1; r < 8; ++r) tmax = fmaxf(tmax, st[0][r]);
    #pragma unroll
    for (int r = 0; r < 8; ++r) tmax = fmaxf(tmax, st[1][r]);
    tmax = fmaxf(tmax, __shfl_xor(tmax, 16, 32));
    const float m_new = fmaxf(m_run, tmax);
    const float alpha = __expf(m_run - m_new);
    float p0[8], p1[8], lsum = 0.0f;
    #pragma unroll
    for (int r = 0; r < 8; ++r) { p0[r] = __expf(st[0][r] - m_new); lsum += p0[r]; }
    #pragma unroll
    for (int r = 0; r < 8; ++r) { p1[r] = __expf(st[1][r] - m_new); lsum += p1[r]; }
    lsum += __shfl_xor(lsum, 16, 32);
    l_run = l_run * alpha + lsum;
    m_run = m_new;
    #pragma unroll
    for (int g = 0; g < 4; ++g)
      #pragma unroll
      for (int r = 0; r < 8; ++r) acc[g][r] *= alpha;

    // ---- P^T B-fragment: RNE + perm-pack, 16 shuffles, perm-select ----
    v16bf pb;
    {
      unsigned int pk[8];
      #pragma unroll
      for (int j = 0; j < 8; ++j)   // bf16(p0[j]) | bf16(p1[j])<<16 via v_perm
        pk[j] = __builtin_amdgcn_perm(rne_bits(p1[j]), rne_bits(p0[j]), 0x07060302u);
      unsigned int lo[8], hi[8];
      #pragma unroll
      for (int j = 0; j < 8; ++j) {
        lo[j] = (unsigned int)__shfl((int)pk[j], n,      32); // rows j
        hi[j] = (unsigned int)__shfl((int)pk[j], n + 16, 32); // rows 8+j
      }
      v8u w;
      #pragma unroll
      for (int k = 0; k < 4; ++k) {
        w[k]     = __builtin_amdgcn_perm(lo[2*k+1], lo[2*k], selw); // u[2k]|u[2k+1]<<16
        w[k + 4] = __builtin_amdgcn_perm(hi[2*k+1], hi[2*k], selw); // u[8+2k]|...
      }
      pb = __builtin_bit_cast(v16bf, w);
    }

    // ---- O^T += V^T x P^T (4 dim-groups) ----
    #pragma unroll
    for (int g = 0; g < 4; ++g)
      acc[g] = __builtin_amdgcn_wmma_f32_16x16x32_bf16(
                   false, __builtin_bit_cast(v16bf, vf[g]), false, pb,
                   (short)0, acc[g], false, false);
  };

  // ---- software-pipelined main loop (2-deep double buffer) ----
  v16us kfA[4], vfA[4], kfB[4], vfB[4];
  loadF(c_lo, kfA, vfA);
  for (int i = 0; i < CPS; i += 2) {
    const int c0 = c_lo + i;
    if (i + 1 < CPS) loadF(c0 + 1, kfB, vfB);
    if (i + 2 < CPS) {  // prefetch two chunks ahead
      __builtin_prefetch(wsK + ((size_t)(h * NTILE + (c0 + 2) * 2) * 2) * 512 + lane * 64, 0, 1);
      __builtin_prefetch(wsV + ((size_t)(h * NCHUNK + (c0 + 2)) * 4) * 512 + lane * 64, 0, 1);
    }
    computeC(c0, kfA, vfA);
    if (i + 1 >= CPS) break;
    if (i + 2 < CPS) loadF(c0 + 2, kfA, vfA);
    computeC(c0 + 1, kfB, vfB);
  }

  // ---- store partials: [block][dim][query] + per-query (m, l) ----
  const size_t pb_i = (size_t)blockIdx.x;
  if (half == 0) {
    wsM[pb_i * 16 + n] = m_run;
    wsL[pb_i * 16 + n] = l_run;
  }
  #pragma unroll
  for (int g = 0; g < 4; ++g)
    #pragma unroll
    for (int r = 0; r < 8; ++r)
      wsO[pb_i * 1024 + (size_t)(g * 16 + r + 8 * half) * 16 + n] = acc[g][r];
}

// ---------------------------------------------------------------------------
// Pass 3: merge NSPLIT partials per (h, qtile) -> y.
// ---------------------------------------------------------------------------
__global__ __launch_bounds__(256)
void reduce_kernel(const float* __restrict__ wsO, const float* __restrict__ wsM,
                   const float* __restrict__ wsL, float* __restrict__ y)
{
  __shared__ float sScale[NSPLIT][16];
  __shared__ float sInvL[16];
  const int b   = blockIdx.x;        // h*32 + qt
  const int h   = b >> 5;
  const int qt  = b & 31;
  const int tid = threadIdx.x;

  if (tid < 16) {
    const int q = tid;
    float mstar = -3.0e38f;
    for (int i = 0; i < NSPLIT; ++i)
      mstar = fmaxf(mstar, wsM[(size_t)(b * NSPLIT + i) * 16 + q]);
    float lstar = 0.0f;
    for (int i = 0; i < NSPLIT; ++i) {
      const float sc = __expf(wsM[(size_t)(b * NSPLIT + i) * 16 + q] - mstar);
      sScale[i][q] = sc;
      lstar += sc * wsL[(size_t)(b * NSPLIT + i) * 16 + q];
    }
    sInvL[q] = 1.0f / lstar;
  }
  __syncthreads();

  #pragma unroll
  for (int k = 0; k < 4; ++k) {
    const int idx = k * 256 + tid;   // d*16 + q
    const int d = idx >> 4, q = idx & 15;
    float o = 0.0f;
    for (int i = 0; i < NSPLIT; ++i)
      o += wsO[(size_t)(b * NSPLIT + i) * 1024 + idx] * sScale[i][q];
    y[((size_t)(qt * 16 + q) * NH + h) * HD + d] = o * sInvL[q];
  }
}

// ---------------------------------------------------------------------------
// Fallback: self-contained single-pass flash attention (no workspace).
// ---------------------------------------------------------------------------
__global__ __launch_bounds__(32)
void attn_kernel(const float* __restrict__ memK, const float* __restrict__ memV,
                 const float* __restrict__ segK, const float* __restrict__ segV,
                 const float* __restrict__ Q, const unsigned char* __restrict__ sos,
                 float* __restrict__ y)
{
  __shared__ float ldsV[32 * 68];
  const int lane = threadIdx.x;
  const int n    = lane & 15;
  const int half = lane >> 4;
  const int h    = blockIdx.x >> 5;
  const int q0   = (blockIdx.x & 31) * 16;
  const float keep  = sos[0] ? 0.0f : 1.0f;
  const float scale = 0.125f;
  const unsigned int selw = half ? 0x07060302u : 0x05040100u;

  v16bf qb[2];
  {
    const float* qr = Q + ((size_t)(q0 + n) * NH + h) * HD;
    #pragma unroll
    for (int c = 0; c < 2; ++c) {
      const int d0 = c * 32 + half * 8;
      v16us u;
      #pragma unroll
      for (int i = 0; i < 4; ++i) {
        float4 v = *(const float4*)(qr + d0 + (i & 1) * 4 + (i >> 1) * 16);
        u[i*4+0] = f2bf(v.x * scale); u[i*4+1] = f2bf(v.y * scale);
        u[i*4+2] = f2bf(v.z * scale); u[i*4+3] = f2bf(v.w * scale);
      }
      qb[c] = __builtin_bit_cast(v16bf, u);
    }
  }

  float m_run = -3.0e38f, l_run = 0.0f;
  v8f acc[4] = { v8f{}, v8f{}, v8f{}, v8f{} };

  for (int kc0 = 0; kc0 < BM + SS; kc0 += 32) {
    const bool  isSeg  = kc0 >= BM;
    const float kscale = isSeg ? 1.0f : keep;

    float st[2][8];
    #pragma unroll
    for (int t = 0; t < 2; ++t) {
      const int key = kc0 + t * 16 + n;
      const float* kr = isSeg ? (segK + ((size_t)(key - BM) * NH + h) * HD)
                              : (memK + ((size_t)key       * NH + h) * HD);
      v8f c = v8f{};
      #pragma unroll
      for (int cc = 0; cc < 2; ++cc) {
        const int d0 = cc * 32 + half * 8;
        float4 a0 = *(const float4*)(kr + d0);
        float4 a1 = *(const float4*)(kr + d0 + 4);
        float4 a2 = *(const float4*)(kr + d0 + 16);
        float4 a3 = *(const float4*)(kr + d0 + 20);
        v16us u;
        u[0]=f2bf(a0.x*kscale); u[1]=f2bf(a0.y*kscale); u[2]=f2bf(a0.z*kscale); u[3]=f2bf(a0.w*kscale);
        u[4]=f2bf(a1.x*kscale); u[5]=f2bf(a1.y*kscale); u[6]=f2bf(a1.z*kscale); u[7]=f2bf(a1.w*kscale);
        u[8]=f2bf(a2.x*kscale); u[9]=f2bf(a2.y*kscale); u[10]=f2bf(a2.z*kscale); u[11]=f2bf(a2.w*kscale);
        u[12]=f2bf(a3.x*kscale); u[13]=f2bf(a3.y*kscale); u[14]=f2bf(a3.z*kscale); u[15]=f2bf(a3.w*kscale);
        c = __builtin_amdgcn_wmma_f32_16x16x32_bf16(
                false, __builtin_bit_cast(v16bf, u), false, qb[cc],
                (short)0, c, false, false);
      }
      #pragma unroll
      for (int r = 0; r < 8; ++r) st[t][r] = c[r];
    }

    if (isSeg) {
      #pragma unroll
      for (int t = 0; t < 2; ++t)
        #pragma unroll
        for (int r = 0; r < 8; ++r) {
          const int j = kc0 + t * 16 + r + 8 * half - BM;
          if (j >= q0 + n) st[t][r] = -3.0e38f;
        }
    }

    float tmax = st[0][0];
    #pragma unroll
    for (int r = 1; r < 8; ++r) tmax = fmaxf(tmax, st[0][r]);
    #pragma unroll
    for (int r = 0; r < 8; ++r) tmax = fmaxf(tmax, st[1][r]);
    tmax = fmaxf(tmax, __shfl_xor(tmax, 16, 32));
    const float m_new = fmaxf(m_run, tmax);
    const float alpha = __expf(m_run - m_new);
    float p0[8], p1[8], lsum = 0.0f;
    #pragma unroll
    for (int r = 0; r < 8; ++r) { p0[r] = __expf(st[0][r] - m_new); lsum += p0[r]; }
    #pragma unroll
    for (int r = 0; r < 8; ++r) { p1[r] = __expf(st[1][r] - m_new); lsum += p1[r]; }
    lsum += __shfl_xor(lsum, 16, 32);
    l_run = l_run * alpha + lsum;
    m_run = m_new;
    #pragma unroll
    for (int g = 0; g < 4; ++g)
      #pragma unroll
      for (int r = 0; r < 8; ++r) acc[g][r] *= alpha;

    v16bf pbf;
    {
      unsigned int pk[8];
      #pragma unroll
      for (int j = 0; j < 8; ++j)
        pk[j] = __builtin_amdgcn_perm(rne_bits(p1[j]), rne_bits(p0[j]), 0x07060302u);
      unsigned int lo[8], hi[8];
      #pragma unroll
      for (int j = 0; j < 8; ++j) {
        lo[j] = (unsigned int)__shfl((int)pk[j], n,      32);
        hi[j] = (unsigned int)__shfl((int)pk[j], n + 16, 32);
      }
      v8u w;
      #pragma unroll
      for (int k = 0; k < 4; ++k) {
        w[k]     = __builtin_amdgcn_perm(lo[2*k+1], lo[2*k], selw);
        w[k + 4] = __builtin_amdgcn_perm(hi[2*k+1], hi[2*k], selw);
      }
      pbf = __builtin_bit_cast(v16bf, w);
    }

    __syncthreads();
    #pragma unroll
    for (int i = 0; i < 16; ++i) {
      const int f  = i * 32 + lane;
      const int kl = f >> 4;
      const int dq = (f & 15) * 4;
      const int key = kc0 + kl;
      const float* vr = isSeg ? (segV + ((size_t)(key - BM) * NH + h) * HD)
                              : (memV + ((size_t)key       * NH + h) * HD);
      float4 v = *(const float4*)(vr + dq);
      v.x *= kscale; v.y *= kscale; v.z *= kscale; v.w *= kscale;
      *(float4*)&ldsV[kl * 68 + dq] = v;
    }
    __syncthreads();

    #pragma unroll
    for (int g = 0; g < 4; ++g) {
      v16us u;
      #pragma unroll
      for (int j = 0; j < 8; ++j) {
        const int k0 = half * 16 + j;
        const int k1 = half * 16 + 8 + j;
        u[j]     = f2bf(ldsV[k0 * 68 + g * 16 + n]);
        u[j + 8] = f2bf(ldsV[k1 * 68 + g * 16 + n]);
      }
      acc[g] = __builtin_amdgcn_wmma_f32_16x16x32_bf16(
                   false, __builtin_bit_cast(v16bf, u), false, pbf,
                   (short)0, acc[g], false, false);
    }
  }

  const float inv = 1.0f / l_run;
  float* yr = y + ((size_t)(q0 + n) * NH + h) * HD;
  #pragma unroll
  for (int g = 0; g < 4; ++g)
    #pragma unroll
    for (int r = 0; r < 8; ++r)
      yr[g * 16 + r + 8 * half] = acc[g][r] * inv;
}

// ---------------------------------------------------------------------------
extern "C" void kernel_launch(void* const* d_in, const int* in_sizes, int n_in,
                              void* d_out, int out_size, void* d_ws, size_t ws_size,
                              hipStream_t stream) {
  const float* memK = (const float*)d_in[0];
  const float* memV = (const float*)d_in[1];
  const float* segK = (const float*)d_in[2];
  const float* segV = (const float*)d_in[3];
  const float* Q    = (const float*)d_in[4];
  const unsigned char* sos = (const unsigned char*)d_in[5];
  const int* wip    = (const int*)d_in[6];

  float* out = (float*)d_out;
  float* y   = out;
  float* nmk = out + (size_t)SS * NH * HD;
  float* nmv = nmk + (size_t)BM * NH * HD;
  float* nwi = nmv + (size_t)BM * NH * HD;

  const long long totalf4 = (long long)BM * NH * HD / 4;
  memupd_kernel<<<(int)((totalf4 + 255) / 256), 256, 0, stream>>>(
      memK, memV, segK, segV, sos, wip, nmk, nmv, nwi);

  const size_t kBytes = (size_t)NH * NTILE * 2 * 512 * sizeof(unsigned short);
  const size_t vBytes = (size_t)NH * NCHUNK * 4 * 512 * sizeof(unsigned short);
  const size_t oBytes = (size_t)NH * 32 * NSPLIT * 1024 * sizeof(float);
  const size_t mBytes = (size_t)NH * 32 * NSPLIT * 16 * sizeof(float);
  const size_t need   = kBytes + vBytes + oBytes + 2 * mBytes;

  if (ws_size >= need) {
    unsigned short* wsK = (unsigned short*)d_ws;
    unsigned short* wsV = (unsigned short*)((char*)d_ws + kBytes);
    float*          wsO = (float*)((char*)d_ws + kBytes + vBytes);
    float*          wsM = (float*)((char*)d_ws + kBytes + vBytes + oBytes);
    float*          wsL = (float*)((char*)d_ws + kBytes + vBytes + oBytes + mBytes);

    prepk_kernel<<<(NH * NTILE * 2 + 7) / 8, 256, 0, stream>>>(memK, segK, sos, wsK);
    prepv_kernel<<<(NH * NCHUNK + 7) / 8,   256, 0, stream>>>(memV, segV, sos, wsV);
    attn_split_kernel<<<NH * 32 * NSPLIT, 32, 0, stream>>>(wsK, wsV, Q, wsO, wsM, wsL);
    reduce_kernel<<<NH * 32, 256, 0, stream>>>(wsO, wsM, wsL, y);
  } else {
    attn_kernel<<<NH * 32, 32, 0, stream>>>(memK, memV, segK, segV, Q, sos, y);
  }
}
